// GSA_45397804319082
// MI455X (gfx1250) — matrix operand order repaired and checked by reading, not compiled, not run on hardware
//
#include <hip/hip_runtime.h>

// ---------------------------------------------------------------------------
// CDNA5 (gfx1250, wave32) lambda-style global self-attention.
// All GEMMs: v_wmma_f32_16x16x32_bf16.  Big GEMMs use 2x4 register tiling
// (32x64 per wave -> 8 WMMAs per 12 b128 loads); 64^3 GEMM families use 2x2;
// the content-context GEMM (K=4096) uses split-K with a deterministic
// fp32 partial reduction.  Softmax / BatchNorm statistics in fp32.
// ---------------------------------------------------------------------------

typedef __attribute__((ext_vector_type(16))) __bf16 bf16x16;
typedef __attribute__((ext_vector_type(8)))  __bf16 bf16x8;
typedef __attribute__((ext_vector_type(8)))  float  f32x8;

// ---- WMMA fragment loaders (per-lane), matching ISA 7.12.2 layouts --------
// A (16x32 bf16): lanes 0-15 -> M=lane, K={0..7,16..23}; lanes 16-31 -> K+8.
// B (32x16 bf16): lanes 0-15 -> N=lane, K={0..7,16..23}; lanes 16-31 -> K+8.
__device__ __forceinline__ bf16x16 frag_a(const __bf16* row, int k0, int kb) {
  union { bf16x16 v; bf16x8 h[2]; } u;
  u.h[0] = *(const bf16x8*)(row + k0 + kb);
  u.h[1] = *(const bf16x8*)(row + k0 + 16 + kb);
  return u.v;
}
__device__ __forceinline__ bf16x16 frag_b(const __bf16* col, int k0, int kb2) {
  union { bf16x16 v; bf16x8 h[2]; } u;
  u.h[0] = *(const bf16x8*)(col + k0 + kb2);
  u.h[1] = *(const bf16x8*)(col + k0 + kb2 + 16);
  return u.v;
}

#define WMMA_BF16(a, b, c) \
  __builtin_amdgcn_wmma_f32_16x16x32_bf16(false, (a), false, (b), (short)0, (c), false, false)

// One wave computes a 32 x (16*NT) f32 block (2 x NT tiles of 16x16).
// A rows K-contiguous with stride lda; B columns K-contiguous with stride ldb.
template <int NT>
__device__ __forceinline__ void wmma_2xN(const __bf16* A, int lda,
                                         const __bf16* B, int ldb,
                                         int kTot, int mn, int hl,
                                         f32x8 acc[2][NT]) {
  const __bf16* ar0 = A + mn * lda;
  const __bf16* ar1 = A + (mn + 16) * lda;
  const __bf16* bc  = B + mn * ldb;
  const int kbA = hl * 8;
  const int kbB = hl * 8;
  for (int k0 = 0; k0 < kTot; k0 += 32) {
    bf16x16 a0 = frag_a(ar0, k0, kbA);
    bf16x16 a1 = frag_a(ar1, k0, kbA);
#pragma unroll
    for (int t = 0; t < NT; ++t) {
      bf16x16 b = frag_b(bc + 16 * t * ldb, k0, kbB);
      acc[0][t] = WMMA_BF16(a0, b, acc[0][t]);
      acc[1][t] = WMMA_BF16(a1, b, acc[1][t]);
    }
  }
}

template <int NT>
__device__ __forceinline__ void zero_acc(f32x8 acc[2][NT]) {
#pragma unroll
  for (int i = 0; i < 2; ++i)
#pragma unroll
    for (int t = 0; t < NT; ++t) acc[i][t] = {};
}

#define LANE_IDS  const int lane = threadIdx.x & 31; const int wv = threadIdx.x >> 5; \
                  const int mn = lane & 15; const int hl = lane >> 4;

// ---- conversions ----------------------------------------------------------
__global__ void k_cvt(const float* __restrict__ s, __bf16* __restrict__ d, int n) {
  int i = blockIdx.x * 256 + threadIdx.x;
  if (i < n) d[i] = (__bf16)s[i];
}
// imgT[b, p(4096), c(256)] = img[b, c, p]
__global__ void k_cvt_imgT(const float* __restrict__ img, __bf16* __restrict__ imgT) {
  int i = blockIdx.x * 256 + threadIdx.x;          // 8*4096*256
  int b = i >> 20, r = i & 1048575;
  int p = r >> 8, c = r & 255;
  imgT[i] = (__bf16)img[(size_t)b * 1048576 + c * 4096 + p];
}
// outT[b, p(4096), c(512)] = out[b, c, p]
__global__ void k_cvt_outT(const float* __restrict__ o, __bf16* __restrict__ oT) {
  int i = blockIdx.x * 256 + threadIdx.x;          // 8*4096*512
  int b = i >> 21, r = i & 2097151;
  int p = r >> 9, c = r & 511;
  oT[i] = (__bf16)o[(size_t)b * 2097152 + c * 4096 + p];
}

// ---- QKV GEMM: [1536x256] x [256x4096] per batch, fused scatter epilogue --
__device__ __forceinline__ void qkv_store(const f32x8& c, int b, int o0, int p,
                                          __bf16* __restrict__ qT, float* __restrict__ kRaw,
                                          __bf16* __restrict__ vB, __bf16* __restrict__ vY) {
  const int x = p >> 6, y = p & 63;
#pragma unroll
  for (int j = 0; j < 8; ++j) {
    int o = o0 + j;
    int part = o >> 9, hh = (o >> 6) & 7, d = o & 63;
    int n = b * 8 + hh;
    float acc = c[j];
    if (part == 0) {
      qT[(size_t)n * 262144 + p * 64 + d] = (__bf16)acc;        // qT[n, m, d]
    } else if (part == 1) {
      kRaw[(size_t)n * 262144 + d * 4096 + p] = acc;            // k fp32 for softmax
    } else {
      __bf16 bv = (__bf16)acc;
      vB[(size_t)n * 262144 + d * 4096 + p] = bv;               // v[n, e, m]
      vY[(size_t)n * 262144 + y * 4096 + d * 64 + x] = bv;      // vY[n, y, e, i=x]
    }
  }
}

__global__ void k_gemm_qkv(const __bf16* __restrict__ W, const __bf16* __restrict__ X,
                           __bf16* __restrict__ qT, float* __restrict__ kRaw,
                           __bf16* __restrict__ vB, __bf16* __restrict__ vY) {
  LANE_IDS
  int job = blockIdx.x * 8 + wv;                   // 8 * 48 * 64 jobs
  int b  = job / (48 * 64);
  int r  = job % (48 * 64);
  int mb = r >> 6, nb = r & 63;
  const __bf16* A = W + mb * 32 * 256;
  const __bf16* B = X + (size_t)b * (4096 * 256) + nb * 64 * 256;
  f32x8 acc[2][4]; zero_acc<4>(acc);
  wmma_2xN<4>(A, 256, B, 256, 256, mn, hl, acc);
  int o0 = mb * 32 + hl * 8;
  int p0 = nb * 64 + mn;
#pragma unroll
  for (int ii = 0; ii < 2; ++ii)
#pragma unroll
    for (int t = 0; t < 4; ++t)
      qkv_store(acc[ii][t], b, o0 + ii * 16, p0 + t * 16, qT, kRaw, vB, vY);
}

// ---- softmax over m=4096 for each (n,d) row -------------------------------
__global__ void k_softmax(const float* __restrict__ kRaw, __bf16* __restrict__ kB) {
  const int row = blockIdx.x, tid = threadIdx.x;
  const float* src = kRaw + (size_t)row * 4096;
  __shared__ float red[256];
  float mx = -3.4e38f;
  for (int j = tid; j < 4096; j += 256) mx = fmaxf(mx, src[j]);
  red[tid] = mx; __syncthreads();
  for (int s = 128; s > 0; s >>= 1) { if (tid < s) red[tid] = fmaxf(red[tid], red[tid + s]); __syncthreads(); }
  const float rowmax = red[0]; __syncthreads();
  float sum = 0.f;
  for (int j = tid; j < 4096; j += 256) sum += __expf(src[j] - rowmax);
  red[tid] = sum; __syncthreads();
  for (int s = 128; s > 0; s >>= 1) { if (tid < s) red[tid] += red[tid + s]; __syncthreads(); }
  const float inv = 1.0f / red[0];
  __bf16* dst = kB + (size_t)row * 4096;
  for (int j = tid; j < 4096; j += 256) dst[j] = (__bf16)(__expf(src[j] - rowmax) * inv);
}

// ---- context[d,e] = sum_m k[d,m] v[e,m]; split-K over 16 chunks of 256 ----
__global__ void k_gemm_ctx_part(const __bf16* __restrict__ kB, const __bf16* __restrict__ vB,
                                float* __restrict__ ctPart) {
  LANE_IDS
  int job = blockIdx.x * 8 + wv;                   // 64 n * 4 sub * 16 kc
  int n = job >> 6, sub = (job >> 4) & 3, kc = job & 15;
  int dt = sub >> 1, et = sub & 1;
  const __bf16* A = kB + (size_t)n * 262144 + dt * 32 * 4096 + kc * 256;
  const __bf16* B = vB + (size_t)n * 262144 + et * 32 * 4096 + kc * 256;
  f32x8 acc[2][2]; zero_acc<2>(acc);
  wmma_2xN<2>(A, 4096, B, 4096, 256, mn, hl, acc);
  int d0 = dt * 32 + hl * 8;
  int e0 = et * 32 + mn;
#pragma unroll
  for (int ii = 0; ii < 2; ++ii)
#pragma unroll
    for (int t = 0; t < 2; ++t)
#pragma unroll
      for (int j = 0; j < 8; ++j)
        ctPart[(((size_t)kc * 64 + n) * 64 + d0 + ii * 16 + j) * 64 + e0 + t * 16] = acc[ii][t][j];
}

// deterministic reduce of 16 partials; store transposed ctB[n, e, d] (bf16)
__global__ void k_ctx_reduce(const float* __restrict__ ctPart, __bf16* __restrict__ ctB) {
  int i = blockIdx.x * 256 + threadIdx.x;          // 64*64*64 (n, d, e)
  int n = i >> 12, d = (i >> 6) & 63, e = i & 63;
  float s = 0.f;
#pragma unroll
  for (int kc = 0; kc < 16; ++kc) s += ctPart[(size_t)kc * 262144 + i];
  ctB[(size_t)n * 4096 + e * 64 + d] = (__bf16)s;
}

// ---- content[e,m] = sum_d ctB[e,d] * qT[m,d]  -> out (fp32, overwrite) ----
__global__ void k_gemm_content(const __bf16* __restrict__ ctB, const __bf16* __restrict__ qT,
                               float* __restrict__ out) {
  LANE_IDS
  int job = blockIdx.x * 8 + wv;                   // 64 * 2 * 64 jobs
  int n = job >> 7, r = job & 127;
  int mb = r >> 6, nb = r & 63;
  const __bf16* A = ctB + (size_t)n * 4096 + mb * 32 * 64;
  const __bf16* B = qT + (size_t)n * 262144 + nb * 64 * 64;
  f32x8 acc[2][4]; zero_acc<4>(acc);
  wmma_2xN<4>(A, 64, B, 64, 64, mn, hl, acc);
  int e0 = mb * 32 + hl * 8;
  int m0 = nb * 64 + mn;
#pragma unroll
  for (int ii = 0; ii < 2; ++ii)
#pragma unroll
    for (int t = 0; t < 4; ++t)
#pragma unroll
      for (int j = 0; j < 8; ++j)
        out[(size_t)n * 262144 + (e0 + ii * 16 + j) * 4096 + m0 + t * 16] = acc[ii][t][j];
}

// ---- Sx[i,y] = sum_d rel_rows[i-x+63,d] q[d,x,y] -> SxB[n,y,x,i] ----------
__global__ void k_gemm_sx(const __bf16* __restrict__ relR, const __bf16* __restrict__ qT,
                          __bf16* __restrict__ SxB) {
  LANE_IDS
  int job = blockIdx.x * 8 + wv;                   // 4096 * 4 jobs
  int prob = job >> 2, sub = job & 3;
  int n = prob >> 6, x = prob & 63;
  int ti = sub >> 1, tj = sub & 1;
  const __bf16* A = relR + (ti * 32 - x + 63) * 64;
  const __bf16* B = qT + (size_t)n * 262144 + (x * 64 + tj * 32) * 64;
  f32x8 acc[2][2]; zero_acc<2>(acc);
  wmma_2xN<2>(A, 64, B, 64, 64, mn, hl, acc);
  int i0 = ti * 32 + hl * 8;
  int y0 = tj * 32 + mn;
#pragma unroll
  for (int ii = 0; ii < 2; ++ii)
#pragma unroll
    for (int t = 0; t < 2; ++t)
#pragma unroll
      for (int j = 0; j < 8; ++j)
        SxB[(((size_t)n * 64 + y0 + t * 16) * 64 + x) * 64 + i0 + ii * 16 + j] = (__bf16)acc[ii][t][j];
}

// ---- Yh[e,x] = sum_i vY[n,y,e,i] SxB[n,y,x,i] -> YhB[n,e,x,y] (raw) -------
__global__ void k_gemm_yh(const __bf16* __restrict__ vY, const __bf16* __restrict__ SxB,
                          __bf16* __restrict__ YhB) {
  LANE_IDS
  int job = blockIdx.x * 8 + wv;                   // 4096 * 4 jobs
  int prob = job >> 2, sub = job & 3;
  int n = prob >> 6, y = prob & 63;
  int ti = sub >> 1, tj = sub & 1;
  const __bf16* A = vY + (size_t)n * 262144 + y * 4096 + ti * 32 * 64;
  const __bf16* B = SxB + (((size_t)n * 64 + y) * 64 + tj * 32) * 64;
  f32x8 acc[2][2]; zero_acc<2>(acc);
  wmma_2xN<2>(A, 64, B, 64, 64, mn, hl, acc);
  int e0 = ti * 32 + hl * 8;
  int x0 = tj * 32 + mn;
#pragma unroll
  for (int ii = 0; ii < 2; ++ii)
#pragma unroll
    for (int t = 0; t < 2; ++t)
#pragma unroll
      for (int j = 0; j < 8; ++j)
        YhB[(size_t)n * 262144 + (e0 + ii * 16 + j) * 4096 + (x0 + t * 16) * 64 + y] = (__bf16)acc[ii][t][j];
}

// ---- BatchNorm stats per channel e over (n,x,y) ---------------------------
__global__ void k_bnstats(const __bf16* __restrict__ YhB, const float* __restrict__ gamma,
                          const float* __restrict__ beta, float* __restrict__ scale,
                          float* __restrict__ shift) {
  const int e = blockIdx.x, tid = threadIdx.x;
  __shared__ float s1[256], s2[256];
  float a = 0.f, b = 0.f;
  for (int j = tid; j < 64 * 4096; j += 256) {
    int n = j >> 12, xy = j & 4095;
    float v = (float)YhB[(size_t)n * 262144 + e * 4096 + xy];
    a += v; b += v * v;
  }
  s1[tid] = a; s2[tid] = b; __syncthreads();
  for (int s = 128; s > 0; s >>= 1) {
    if (tid < s) { s1[tid] += s1[tid + s]; s2[tid] += s2[tid + s]; }
    __syncthreads();
  }
  if (tid == 0) {
    const float N = 262144.f;
    float mean = s1[0] / N;
    float var  = s2[0] / N - mean * mean;
    float sc   = gamma[e] * rsqrtf(var + 1e-5f);
    scale[e] = sc;
    shift[e] = beta[e] - mean * sc;
  }
}

__global__ void k_bn_apply(__bf16* __restrict__ YhB, const float* __restrict__ scale,
                           const float* __restrict__ shift) {
  int i = blockIdx.x * 256 + threadIdx.x;          // 64*64*4096
  int e = (i >> 12) & 63;
  YhB[i] = (__bf16)((float)YhB[i] * scale[e] + shift[e]);
}

// ---- Sy[i,x] = sum_d rel_cols[i-y+63,d] q[d,x,y] -> SyB[n,x,y,i] ----------
__global__ void k_gemm_sy(const __bf16* __restrict__ relC, const __bf16* __restrict__ qT,
                          __bf16* __restrict__ SyB) {
  LANE_IDS
  int job = blockIdx.x * 8 + wv;                   // 4096 * 4 jobs
  int prob = job >> 2, sub = job & 3;
  int n = prob >> 6, y = prob & 63;
  int ti = sub >> 1, tj = sub & 1;
  const __bf16* A = relC + (ti * 32 - y + 63) * 64;
  const __bf16* B = qT + (size_t)n * 262144 + y * 64 + tj * 32 * 4096;
  f32x8 acc[2][2]; zero_acc<2>(acc);
  wmma_2xN<2>(A, 64, B, 4096, 64, mn, hl, acc);
  int i0 = ti * 32 + hl * 8;
  int x0 = tj * 32 + mn;
#pragma unroll
  for (int ii = 0; ii < 2; ++ii)
#pragma unroll
    for (int t = 0; t < 2; ++t)
#pragma unroll
      for (int j = 0; j < 8; ++j)
        SyB[(((size_t)n * 64 + x0 + t * 16) * 64 + y) * 64 + i0 + ii * 16 + j] = (__bf16)acc[ii][t][j];
}

// ---- rel[e,y] = sum_i YhB[n,e,x,i] SyB[n,x,y,i] -> out += ------------------
__global__ void k_gemm_rel(const __bf16* __restrict__ YhB, const __bf16* __restrict__ SyB,
                           float* __restrict__ out) {
  LANE_IDS
  int job = blockIdx.x * 8 + wv;                   // 4096 * 4 jobs
  int prob = job >> 2, sub = job & 3;
  int n = prob >> 6, x = prob & 63;
  int ti = sub >> 1, tj = sub & 1;
  const __bf16* A = YhB + (size_t)n * 262144 + ti * 32 * 4096 + x * 64;
  const __bf16* B = SyB + (((size_t)n * 64 + x) * 64 + tj * 32) * 64;
  f32x8 acc[2][2]; zero_acc<2>(acc);
  wmma_2xN<2>(A, 4096, B, 64, 64, mn, hl, acc);
  int e0 = ti * 32 + hl * 8;
  int y0 = tj * 32 + mn;
#pragma unroll
  for (int ii = 0; ii < 2; ++ii)
#pragma unroll
    for (int t = 0; t < 2; ++t)
#pragma unroll
      for (int j = 0; j < 8; ++j)
        out[(size_t)n * 262144 + (e0 + ii * 16 + j) * 4096 + x * 64 + y0 + t * 16] += acc[ii][t][j];
}

// ---- final 1x1 conv: [256x512] x [512x4096] per batch + bias --------------
__global__ void k_gemm_final(const __bf16* __restrict__ Wo, const __bf16* __restrict__ outT,
                             const float* __restrict__ bias, float* __restrict__ dout) {
  LANE_IDS
  int job = blockIdx.x * 8 + wv;                   // 8 * 8 * 64 jobs
  int b = job >> 9, r = job & 511;
  int mb = r >> 6, nb = r & 63;
  const __bf16* A = Wo + mb * 32 * 512;
  const __bf16* B = outT + (size_t)b * 2097152 + nb * 64 * 512;
  f32x8 acc[2][4]; zero_acc<4>(acc);
  wmma_2xN<4>(A, 512, B, 512, 512, mn, hl, acc);
  int o0 = mb * 32 + hl * 8;
  int p0 = nb * 64 + mn;
#pragma unroll
  for (int ii = 0; ii < 2; ++ii)
#pragma unroll
    for (int t = 0; t < 4; ++t)
#pragma unroll
      for (int j = 0; j < 8; ++j) {
        int o = o0 + ii * 16 + j;
        dout[(size_t)b * 1048576 + o * 4096 + p0 + t * 16] = acc[ii][t][j] + bias[o];
      }
}

// ---------------------------------------------------------------------------
extern "C" void kernel_launch(void* const* d_in, const int* in_sizes, int n_in,
                              void* d_out, int out_size, void* d_ws, size_t ws_size,
                              hipStream_t stream) {
  const float* img    = (const float*)d_in[0];
  const float* w_qkv  = (const float*)d_in[1];
  const float* w_out  = (const float*)d_in[2];
  const float* b_out  = (const float*)d_in[3];
  const float* relR_f = (const float*)d_in[4];
  const float* relC_f = (const float*)d_in[5];
  const float* gamma  = (const float*)d_in[6];
  const float* beta   = (const float*)d_in[7];
  float* dout = (float*)d_out;

  // workspace carve-up (256-byte aligned)
  char* base = (char*)d_ws;
  size_t off = 0;
  auto alloc = [&](size_t bytes) -> char* {
    char* p = base + off;
    off += (bytes + 255) & ~(size_t)255;
    return p;
  };
  __bf16* imgT   = (__bf16*)alloc((size_t)8388608 * 2);
  __bf16* wqkvB  = (__bf16*)alloc((size_t)393216 * 2);
  __bf16* woutB  = (__bf16*)alloc((size_t)131072 * 2);
  __bf16* relRB  = (__bf16*)alloc((size_t)8128 * 2);
  __bf16* relCB  = (__bf16*)alloc((size_t)8128 * 2);
  __bf16* qT     = (__bf16*)alloc((size_t)16777216 * 2);
  float*  kRaw   = (float*) alloc((size_t)16777216 * 4);
  __bf16* kB     = (__bf16*)alloc((size_t)16777216 * 2);
  __bf16* vB     = (__bf16*)alloc((size_t)16777216 * 2);
  __bf16* vY     = (__bf16*)alloc((size_t)16777216 * 2);
  float*  ctPart = (float*) alloc((size_t)16 * 262144 * 4);
  __bf16* ctB    = (__bf16*)alloc((size_t)262144 * 2);
  __bf16* SxB    = (__bf16*)alloc((size_t)16777216 * 2);
  __bf16* SyB    = (__bf16*)alloc((size_t)16777216 * 2);
  __bf16* YhB    = (__bf16*)alloc((size_t)16777216 * 2);
  float*  outF   = (float*) alloc((size_t)16777216 * 4);
  __bf16* outT   = (__bf16*)alloc((size_t)16777216 * 2);
  float*  bnSc   = (float*) alloc(64 * 4);
  float*  bnSh   = (float*) alloc(64 * 4);

  // 1) precision conversions
  k_cvt<<<1536, 256, 0, stream>>>(w_qkv, wqkvB, 393216);
  k_cvt<<<512, 256, 0, stream>>>(w_out, woutB, 131072);
  k_cvt<<<32, 256, 0, stream>>>(relR_f, relRB, 8128);
  k_cvt<<<32, 256, 0, stream>>>(relC_f, relCB, 8128);
  k_cvt_imgT<<<32768, 256, 0, stream>>>(img, imgT);

  // 2) QKV projection (WMMA, 2x4) with fused q/k/v scatter
  k_gemm_qkv<<<3072, 256, 0, stream>>>(wqkvB, imgT, qT, kRaw, vB, vY);

  // 3) content path
  k_softmax<<<4096, 256, 0, stream>>>(kRaw, kB);
  k_gemm_ctx_part<<<512, 256, 0, stream>>>(kB, vB, ctPart);
  k_ctx_reduce<<<1024, 256, 0, stream>>>(ctPart, ctB);
  k_gemm_content<<<1024, 256, 0, stream>>>(ctB, qT, outF);

  // 4) relative-position path
  k_gemm_sx<<<2048, 256, 0, stream>>>(relRB, qT, SxB);
  k_gemm_yh<<<2048, 256, 0, stream>>>(vY, SxB, YhB);
  k_bnstats<<<64, 256, 0, stream>>>(YhB, gamma, beta, bnSc, bnSh);
  k_bn_apply<<<65536, 256, 0, stream>>>(YhB, bnSc, bnSh);
  k_gemm_sy<<<2048, 256, 0, stream>>>(relCB, qT, SyB);
  k_gemm_rel<<<2048, 256, 0, stream>>>(YhB, SyB, outF);

  // 5) output projection (2x4)
  k_cvt_outT<<<65536, 256, 0, stream>>>(outF, outT);
  k_gemm_final<<<512, 256, 0, stream>>>(woutB, outT, b_out, dout);
}
